// MultiHeadAttention_46213848104966
// MI455X (gfx1250) — matrix operand order repaired and checked by reading, not compiled, not run on hardware
//
#include <hip/hip_runtime.h>

// MI455X / gfx1250, wave32. Fused flash-attention MHA with f16 WMMA and
// gfx1250 async global->LDS staging with double buffering.
// B=4, S=2048, D=1024, H=16, DK=10 (pad 16/32), DV=12 (pad 16).

typedef __attribute__((ext_vector_type(16))) _Float16 v16h;
typedef __attribute__((ext_vector_type(8)))  float    v8f;
typedef __attribute__((ext_vector_type(4)))  int      v4i;

#if defined(__has_builtin)
#if __has_builtin(__builtin_amdgcn_global_load_async_to_lds_b128) && \
    __has_builtin(__builtin_amdgcn_s_wait_asynccnt)
#define USE_ASYNC_LDS 1
#endif
#endif
#ifndef USE_ASYNC_LDS
#define USE_ASYNC_LDS 0
#endif

union AF {
  v16h v;
  _Float16 h[16];
  unsigned u[8];
};

// 16-bit A operand k order chosen by this code (not the ISA table order): element e lives in VGPR e>>1,
// half e&1. K = 2*vg + 16*(lane>>4) + (e&1) (contiguous, matches B).
__device__ __forceinline__ int a_kbase(int vg, int g) {
  return vg * 2 + 16 * g;
}

__device__ __forceinline__ v8f wmma_f16(v16h a, v16h b, v8f c) {
  return __builtin_amdgcn_wmma_f32_16x16x32_f16(false, a, false, b, (short)0,
                                                c, false, false);
}

// 16-byte global -> LDS copy: async DMA (tracked by ASYNCcnt) on gfx1250.
// Builtin signature (from hipcc diagnostic): param0 = AS1 v4i* (global src).
__device__ __forceinline__ void copy16_g2l(const _Float16* g, _Float16* l) {
#if USE_ASYNC_LDS
  __builtin_amdgcn_global_load_async_to_lds_b128(
      (__attribute__((address_space(1))) v4i*)g,
      (__attribute__((address_space(3))) v4i*)l, 0, 0);
#else
  *(uint4*)l = *(const uint4*)g;
#endif
}

__device__ __forceinline__ void wait_async_lds() {
#if USE_ASYNC_LDS
  __builtin_amdgcn_s_wait_asynccnt(0);
#endif
}

// ---------------------------------------------------------------------------
// Kernel 0: weight prep. WQt/WKt/WVt[256][1024] half (n = h*16+j, zero pad),
// WOt[1024][192] half (transposed WO).
// ---------------------------------------------------------------------------
__global__ void prep_kernel(const float* __restrict__ WQ,
                            const float* __restrict__ WK,
                            const float* __restrict__ WV,
                            const float* __restrict__ WO,
                            _Float16* __restrict__ WQt,
                            _Float16* __restrict__ WKt,
                            _Float16* __restrict__ WVt,
                            _Float16* __restrict__ WOt) {
  const int stride = gridDim.x * blockDim.x;
  int idx = blockIdx.x * blockDim.x + threadIdx.x;
  const int total1 = 256 * 1024;
  for (int i = idx; i < total1; i += stride) {
    int n = i >> 10;          // 0..255
    int d = i & 1023;
    int h = n >> 4, j = n & 15;
    WQt[i] = (j < 10) ? (_Float16)WQ[((size_t)h * 1024 + d) * 10 + j]
                      : (_Float16)0.f;
    WKt[i] = (j < 10) ? (_Float16)WK[((size_t)h * 1024 + d) * 10 + j]
                      : (_Float16)0.f;
    WVt[i] = (j < 12) ? (_Float16)WV[((size_t)h * 1024 + d) * 12 + j]
                      : (_Float16)0.f;
  }
  const int total2 = 1024 * 192;
  for (int i = idx; i < total2; i += stride) {
    int d = i / 192, j = i % 192;
    WOt[i] = (_Float16)WO[(size_t)j * 1024 + d];
  }
}

// ---------------------------------------------------------------------------
// Kernel 1: projection GEMM. X[8192,1024] fp32 x Wt[256,1024] half ->
//   mode 0/1 (q/k): out[b*16+h][s][16] half
//   mode 2   (v)  : out[b*16+h][dv(16)][2048] half (transposed for P*V)
// One wave = 16 rows x 64 cols, K-loop 32 x WMMA 16x16x32.
// ---------------------------------------------------------------------------
__global__ __launch_bounds__(128) void proj_kernel(
    const float* __restrict__ X, const _Float16* __restrict__ Wt,
    _Float16* __restrict__ out, int mode) {
  const int lane = threadIdx.x & 31;
  const int wave = threadIdx.x >> 5;
  const int wg = blockIdx.x * 4 + wave;       // 0..2047
  const int mtile = wg >> 2;                  // 0..511
  const int n0 = (wg & 3) * 64;               // 0,64,128,192
  const int g = lane >> 4, ln = lane & 15;
  const int m0 = mtile * 16;

  v8f acc[4];
  const v8f zero = {0.f, 0.f, 0.f, 0.f, 0.f, 0.f, 0.f, 0.f};
#pragma unroll
  for (int t = 0; t < 4; ++t) acc[t] = zero;

  const float* xrow = X + (size_t)(m0 + ln) * 1024;
  for (int kb = 0; kb < 32; ++kb) {
    AF a;
#pragma unroll
    for (int vg = 0; vg < 8; ++vg) {
      int K = a_kbase(vg, g);
      const float* p = xrow + kb * 32 + K;    // K even -> 8B aligned pair
      a.h[2 * vg]     = (_Float16)p[0];
      a.h[2 * vg + 1] = (_Float16)p[1];
    }
#pragma unroll
    for (int t = 0; t < 4; ++t) {
      // B layout: lane = col N, K = g*16 + e, contiguous 32B
      const _Float16* bp =
          Wt + (size_t)(n0 + t * 16 + ln) * 1024 + kb * 32 + g * 16;
      v16h bv = *(const v16h*)bp;
      acc[t] = wmma_f16(a.v, bv, acc[t]);
    }
  }

#pragma unroll
  for (int t = 0; t < 4; ++t) {
    int n = n0 + t * 16 + ln;
    int h = n >> 4, j = n & 15;
#pragma unroll
    for (int r = 0; r < 8; ++r) {
      int m = m0 + r + 8 * g;                 // C layout: M = r + 8*g
      int b = m >> 11, s = m & 2047;
      _Float16 val = (_Float16)acc[t][r];
      if (mode == 2)
        out[(((size_t)(b * 16 + h) * 16 + j) * 2048) + s] = val;
      else
        out[(((size_t)(b * 16 + h) * 2048 + s) * 16) + j] = val;
    }
  }
}

// ---------------------------------------------------------------------------
// Kernel 2: fused flash attention. 4 waves/block share one (b,h); wave w owns
// one 16-row q tile. Per 32-key block: K/V tiles are staged cooperatively
// into LDS (async DMA, double-buffered so the next block's DMA overlaps this
// block's WMMAs), then 2 score WMMAs (K padded 10->32 with register zeros),
// fp32 online softmax, P transposed via per-wave LDS slice, 1 P*V WMMA.
// ---------------------------------------------------------------------------
__global__ __launch_bounds__(128) void attn_kernel(
    const _Float16* __restrict__ q, const _Float16* __restrict__ k,
    const _Float16* __restrict__ vT, _Float16* __restrict__ concat) {
  __shared__ __align__(32) _Float16 ldsK[2][32 * 16];  // [buf][key*16 + f]
  __shared__ __align__(32) _Float16 ldsV[2][16 * 32];  // [buf][dv*32 + key]
  __shared__ __align__(32) _Float16 ldsP[4][512];      // per-wave 16x32 P

  const int tid = threadIdx.x;
  const int lane = tid & 31;
  const int wave = tid >> 5;
  const int bh = blockIdx.x >> 5;                       // 0..63 (b*16+h)
  const int stile = ((blockIdx.x & 31) << 2) | wave;    // 0..127
  const int b = bh >> 4, h = bh & 15;
  const int g = lane >> 4, ln = lane & 15;
  const int s0 = stile * 16;

  const _Float16* kbh = k + (size_t)bh * 2048 * 16;
  const _Float16* vbh = vT + (size_t)bh * 16 * 2048;

  // Load loop-invariant A(q) tile: row = ln, features padded to 32 (K>=16 -> 0)
  AF aq;
  const _Float16* qrow = q + ((size_t)bh * 2048 + s0 + ln) * 16;
#pragma unroll
  for (int vg = 0; vg < 8; ++vg) {
    int K = a_kbase(vg, g);
    if (K < 16) {
      aq.h[2 * vg]     = qrow[K];
      aq.h[2 * vg + 1] = qrow[K + 1];
    } else {
      aq.h[2 * vg] = (_Float16)0.f;
      aq.h[2 * vg + 1] = (_Float16)0.f;
    }
  }

  const v8f zero = {0.f, 0.f, 0.f, 0.f, 0.f, 0.f, 0.f, 0.f};
  v8f accv = zero;
  float mrun[8], lrun[8];
#pragma unroll
  for (int r = 0; r < 8; ++r) { mrun[r] = -1e30f; lrun[r] = 0.f; }

  const float scale = 0.31622776601683794f;   // 1/sqrt(DK=10)
  const unsigned* lp = (const unsigned*)&ldsP[wave][0];

  // Cooperative K/V tile stage: waves 0-1 copy the 1KB K tile (64 x 16B
  // chunks), waves 2-3 copy the 1KB V tile. Wave-uniform split keeps EXEC
  // full for each async instruction.
  auto issue_tiles = [&](int t0, int buf) {
    if (tid < 64) {
      int row = tid >> 1, off = (tid & 1) * 8;
      copy16_g2l(kbh + (size_t)(t0 + row) * 16 + off,
                 &ldsK[buf][row * 16 + off]);
    } else {
      int c = tid - 64;
      int dv = c >> 2, off = (c & 3) * 8;
      copy16_g2l(vbh + (size_t)dv * 2048 + t0 + off,
                 &ldsV[buf][dv * 32 + off]);
    }
  };

  issue_tiles(0, 0);
  wait_async_lds();
  __syncthreads();

  for (int it = 0; it < 64; ++it) {
    const int cur = it & 1, nxt = cur ^ 1;
    const int t0 = it * 32;
    if (it + 1 < 64) issue_tiles(t0 + 32, nxt);   // overlap DMA with compute

    // B(k^T) from LDS: lane = key col; K(feature) = g*16+e -> g==1 pad zeros
    v16h bk0, bk1;
    if (g == 0) {
      bk0 = *(const v16h*)(&ldsK[cur][ln * 16]);
      bk1 = *(const v16h*)(&ldsK[cur][(16 + ln) * 16]);
    } else {
      AF z;
#pragma unroll
      for (int i = 0; i < 8; ++i) z.u[i] = 0u;
      bk0 = z.v; bk1 = z.v;
    }
    v8f sv0 = wmma_f16(aq.v, bk0, zero);      // scores, keys t0..t0+15
    v8f sv1 = wmma_f16(aq.v, bk1, zero);      // scores, keys t0+16..t0+31

    // Online softmax per row (row M = r + 8*g; N spread over 16 lanes)
#pragma unroll
    for (int r = 0; r < 8; ++r) {
      float x0 = sv0[r] * scale, x1 = sv1[r] * scale;
      float tm = fmaxf(x0, x1);
      tm = fmaxf(tm, __shfl_xor(tm, 1, 32));
      tm = fmaxf(tm, __shfl_xor(tm, 2, 32));
      tm = fmaxf(tm, __shfl_xor(tm, 4, 32));
      tm = fmaxf(tm, __shfl_xor(tm, 8, 32));
      float mnew = fmaxf(mrun[r], tm);
      float f = __expf(mrun[r] - mnew);
      float e0 = __expf(x0 - mnew), e1 = __expf(x1 - mnew);
      float rs = e0 + e1;
      rs += __shfl_xor(rs, 1, 32);
      rs += __shfl_xor(rs, 2, 32);
      rs += __shfl_xor(rs, 4, 32);
      rs += __shfl_xor(rs, 8, 32);
      lrun[r] = lrun[r] * f + rs;
      mrun[r] = mnew;
      accv[r] = accv[r] * f;
      ldsP[wave][(r + 8 * g) * 32 + ln]      = (_Float16)e0;
      ldsP[wave][(r + 8 * g) * 32 + 16 + ln] = (_Float16)e1;
    }
    __syncthreads();  // order P writes vs. A-layout re-reads

    // Re-read P in A layout: row = ln, K = keys 0..31; pairs are contiguous
    AF ap;
#pragma unroll
    for (int vg = 0; vg < 8; ++vg) {
      int K = a_kbase(vg, g);
      ap.u[vg] = lp[(ln * 32 + K) >> 1];
    }
    // B(v) from LDS: lane = dv col; K = key = g*16+e, contiguous 32B
    v16h bv = *(const v16h*)(&ldsV[cur][ln * 32 + g * 16]);
    accv = wmma_f16(ap.v, bv, accv);

    wait_async_lds();   // next-block tiles resident in LDS
    __syncthreads();    // publish to all waves; guard cur-buffer reuse
  }

  // Normalize and store concat[b][s][h*12+dv] (dv<12 only)
#pragma unroll
  for (int r = 0; r < 8; ++r) {
    if (ln < 12) {
      int s = s0 + r + 8 * g;
      float val = accv[r] / lrun[r];
      concat[((size_t)b * 2048 + s) * 192 + h * 12 + ln] = (_Float16)val;
    }
  }
}

// ---------------------------------------------------------------------------
// Kernel 3: output projection. concat[8192,192] x WOt[1024,192]^T -> out fp32.
// K = 192 = exactly 6 WMMA steps. Wave = 16 rows x 64 cols.
// ---------------------------------------------------------------------------
__global__ __launch_bounds__(128) void outproj_kernel(
    const _Float16* __restrict__ concat, const _Float16* __restrict__ WOt,
    float* __restrict__ out) {
  const int lane = threadIdx.x & 31;
  const int wave = threadIdx.x >> 5;
  const int wg = blockIdx.x * 4 + wave;       // 0..8191
  const int mtile = wg >> 4;                  // 0..511
  const int n0 = (wg & 15) * 64;              // 0..960
  const int g = lane >> 4, ln = lane & 15;
  const int m0 = mtile * 16;

  v8f acc[4];
  const v8f zero = {0.f, 0.f, 0.f, 0.f, 0.f, 0.f, 0.f, 0.f};
#pragma unroll
  for (int t = 0; t < 4; ++t) acc[t] = zero;

  const unsigned* arow = (const unsigned*)(concat + (size_t)(m0 + ln) * 192);
#pragma unroll
  for (int kb = 0; kb < 6; ++kb) {
    AF a;
#pragma unroll
    for (int vg = 0; vg < 8; ++vg) {
      int K = a_kbase(vg, g);
      a.u[vg] = arow[(kb * 32 + K) >> 1];
    }
#pragma unroll
    for (int t = 0; t < 4; ++t) {
      const _Float16* bp =
          WOt + (size_t)(n0 + t * 16 + ln) * 192 + kb * 32 + g * 16;
      v16h bv = *(const v16h*)bp;
      acc[t] = wmma_f16(a.v, bv, acc[t]);
    }
  }

#pragma unroll
  for (int t = 0; t < 4; ++t) {
    int d = n0 + t * 16 + ln;
#pragma unroll
    for (int r = 0; r < 8; ++r) {
      int m = m0 + r + 8 * g;
      out[(size_t)m * 1024 + d] = acc[t][r];
    }
  }
}

// ---------------------------------------------------------------------------
// Host launcher. Inputs: Q,K,V [4,2048,1024] f32; WQ,WK [16,1024,10];
// WV [16,1024,12]; WO [192,1024]. Output [4,2048,1024] f32.
// ---------------------------------------------------------------------------
extern "C" void kernel_launch(void* const* d_in, const int* in_sizes, int n_in,
                              void* d_out, int out_size, void* d_ws,
                              size_t ws_size, hipStream_t stream) {
  const float* Q  = (const float*)d_in[0];
  const float* K  = (const float*)d_in[1];
  const float* V  = (const float*)d_in[2];
  const float* WQ = (const float*)d_in[3];
  const float* WK = (const float*)d_in[4];
  const float* WV = (const float*)d_in[5];
  const float* WO = (const float*)d_in[6];
  float* out = (float*)d_out;

  char* ws = (char*)d_ws;
  _Float16* WQt = (_Float16*)(ws + (size_t)0);            // 512 KB
  _Float16* WKt = (_Float16*)(ws + ((size_t)512 << 10));  // 512 KB
  _Float16* WVt = (_Float16*)(ws + ((size_t)1024 << 10)); // 512 KB
  _Float16* WOt = (_Float16*)(ws + ((size_t)1536 << 10)); // 384 KB
  _Float16* qws = (_Float16*)(ws + ((size_t)2048 << 10)); // 4 MB  [bh][s][16]
  _Float16* kws = (_Float16*)(ws + ((size_t)6144 << 10)); // 4 MB  [bh][s][16]
  _Float16* vT  = (_Float16*)(ws + ((size_t)10240 << 10));// 4 MB  [bh][16][s]
  _Float16* cc  = (_Float16*)(ws + ((size_t)14336 << 10));// 3 MB  [bs][192]

  prep_kernel<<<1024, 256, 0, stream>>>(WQ, WK, WV, WO, WQt, WKt, WVt, WOt);
  proj_kernel<<<512, 128, 0, stream>>>(Q, WQt, qws, 0);
  proj_kernel<<<512, 128, 0, stream>>>(K, WKt, kws, 1);
  proj_kernel<<<512, 128, 0, stream>>>(V, WVt, vT, 2);
  attn_kernel<<<2048, 128, 0, stream>>>(qws, kws, vT, cc);
  outproj_kernel<<<2048, 128, 0, stream>>>(cc, WOt, out);
}
